// Pointy_81037442941376
// MI455X (gfx1250) — matrix operand order repaired and checked by reading, not compiled, not run on hardware
//
#include <hip/hip_runtime.h>
#include <hip/hip_bf16.h>

typedef __attribute__((ext_vector_type(16))) _Float16 v16h;
typedef __attribute__((ext_vector_type(2)))  _Float16 v2h;
typedef __attribute__((ext_vector_type(2)))  __fp16   v2fp16;
typedef __attribute__((ext_vector_type(8)))  float    v8f;

#define POINTS 16384
#define NBATCH 8
#define NANCH  16
#define NC1    16   // chunks for ae phase
#define NC3    4    // chunks per (b,anchor) for pe phase

union V16U { v16h v; unsigned u[8]; };
union V8FU { v8f  v; float    f[8]; };

static __device__ __forceinline__ unsigned pkh(float a, float b) {
  v2fp16 h = __builtin_amdgcn_cvt_pkrtz(a, b);
  return __builtin_bit_cast(unsigned, h);
}

// packed f16 leaky-relu: max(h, 0.01*h) -> v_pk_mul_f16 + v_pk_max_num_f16
static __device__ __forceinline__ unsigned lrelu_pk(unsigned u) {
  const v2h c01 = { (_Float16)0.01f, (_Float16)0.01f };
  v2h h = __builtin_bit_cast(v2h, u);
  h = __builtin_elementwise_max(h, h * c01);
  return __builtin_bit_cast(unsigned, h);
}

// lane <-> lane^16 exchange as pure VALU (v_permlanex16_b32, identity selects).
// Must be called unconditionally (full EXEC) — callers pre-select the payload.
static __device__ __forceinline__ unsigned xor16u(unsigned v) {
  return __builtin_amdgcn_permlanex16(v, v, 0x76543210u, 0xfedcba98u, false, false);
}
static __device__ __forceinline__ float xor16f(float v) {
  return __builtin_bit_cast(float, xor16u(__builtin_bit_cast(unsigned, v)));
}

// Weight tile in WMMA-A layout: rows = out-features m0..m0+15, K = 32 in-features.
// K order used here (contiguous, B tiles match): lane<16 -> K 0-7 (v0..3 pairs) & 8-15 (v4..7);
//                     lane>=16 -> K 16-23 & 24-31. Row M = lane&15.
static __device__ __forceinline__ v16h load_A_tile(const float* __restrict__ W,
                                                   int m0, int lane) {
  const float* wr = W + (m0 + (lane & 15)) * 32;
  const int kb0 = (lane & 16) ? 16 : 0;
  const int kb1 = (lane & 16) ? 24 : 8;
  V16U a;
#pragma unroll
  for (int j = 0; j < 4; ++j) {
    a.u[j]     = pkh(wr[kb0 + 2*j], wr[kb0 + 2*j + 1]);
    a.u[4 + j] = pkh(wr[kb1 + 2*j], wr[kb1 + 2*j + 1]);
  }
  return a.v;
}

// Bias replicated per row in the C/D layout (lane = point col, VGPR r = feat m0+r / m0+8+r)
static __device__ __forceinline__ v8f bias_C(const float* __restrict__ B, int m0, int lane) {
  const int base = m0 + ((lane & 16) ? 8 : 0);
  V8FU c;
#pragma unroll
  for (int r = 0; r < 8; ++r) c.f[r] = B[base + r];
  return c.v;
}

static __device__ __forceinline__ v8f wmma_f16(v16h a, v16h b, v8f c) {
  return __builtin_amdgcn_wmma_f32_16x16x32_f16(false, a, false, b, (short)0, c, false, false);
}

// 3-layer MLP [2->32->32->32] over a point stream, sum-reduced over points.
// ROT=true applies anchor translate+rotate first (pe phase).
template<bool ROT, int NCHUNK>
__global__ void __launch_bounds__(128, 1)
mlp3_kernel(const float2* __restrict__ x,
            const float* __restrict__ W1, const float* __restrict__ B1,
            const float* __restrict__ W2, const float* __restrict__ B2,
            const float* __restrict__ W3,
            const float4* __restrict__ anc,
            float* __restrict__ partial)
{
  const int lane = threadIdx.x & 31;
  const int wave = threadIdx.x >> 5;
  constexpr int per_b = (ROT ? NANCH : 1) * NCHUNK;
  const int b     = blockIdx.x / per_b;
  const int rem   = blockIdx.x % per_b;
  const int a     = ROT ? (rem / NCHUNK) : 0;
  const int chunk = rem % NCHUNK;
  constexpr int ppw   = POINTS / NCHUNK;   // points per workgroup
  constexpr int iters = ppw / 128;         // 4 waves * 32 points

  float px = 0.f, py = 0.f, rc = 1.f, rs = 0.f;
  if (ROT) { float4 A = anc[b * NANCH + a]; px = A.x; py = A.y; rc = A.z; rs = A.w; }

  // layer-1 weights, packed f16 pairs (features 2j, 2j+1); uniform -> SGPRs
  unsigned w0p[16], w1p[16], bp[16];
#pragma unroll
  for (int j = 0; j < 16; ++j) {
    w0p[j] = pkh(W1[(2*j)*2 + 0], W1[(2*j+1)*2 + 0]);
    w1p[j] = pkh(W1[(2*j)*2 + 1], W1[(2*j+1)*2 + 1]);
    bp[j]  = pkh(B1[2*j], B1[2*j+1]);
  }
  // layer-2/3 weights as resident WMMA A tiles
  const v16h A2lo = load_A_tile(W2, 0, lane);
  const v16h A2hi = load_A_tile(W2, 16, lane);
  const v16h A3lo = load_A_tile(W3, 0, lane);
  const v16h A3hi = load_A_tile(W3, 16, lane);
  const v8f  cb_lo = bias_C(B2, 0, lane);
  const v8f  cb_hi = bias_C(B2, 16, lane);

  v8f acc_lo = {0,0,0,0,0,0,0,0};
  v8f acc_hi = {0,0,0,0,0,0,0,0};
  const bool lo = (lane & 16) == 0;

  const int p0 = chunk * ppw + wave * (ppw / 4);
  const float2* xb = x + b * POINTS;

  // layer2 -> relayout -> layer3 for one 16-point B tile (low register pressure)
  auto process_half = [&](v16h bt) {
    v8f dlo = wmma_f16(A2lo, bt, cb_lo);
    v8f dhi = wmma_f16(A2hi, bt, cb_hi);
    V8FU elo, ehi; elo.v = dlo; ehi.v = dhi;
    V16U bn;
#pragma unroll
    for (int j = 0; j < 4; ++j) {
      float s0 = lo ? ehi.f[2*j]     : elo.f[2*j];
      float s1 = lo ? ehi.f[2*j + 1] : elo.f[2*j + 1];
      float g0 = xor16f(s0);
      float g1 = xor16f(s1);
      float a0 = lo ? elo.f[2*j]     : g0;   // K-low  elems 2j, 2j+1
      float a1 = lo ? elo.f[2*j + 1] : g1;
      float c0 = lo ? g0 : ehi.f[2*j];       // K-high elems 2j, 2j+1
      float c1 = lo ? g1 : ehi.f[2*j + 1];
      bn.u[j]     = lrelu_pk(pkh(a0, a1));
      bn.u[4 + j] = lrelu_pk(pkh(c0, c1));
    }
    acc_lo = wmma_f16(A3lo, bn.v, acc_lo);
    acc_hi = wmma_f16(A3hi, bn.v, acc_hi);
  };

  for (int it = 0; it < iters; ++it) {
    const int p = p0 + it * 32 + lane;
    if (it + 1 < iters) __builtin_prefetch(&xb[p + 32], 0, 1);
    float2 xp = xb[p];
    float r0, r1;
    if (ROT) {
      float d0 = xp.x - px, d1 = xp.y - py;
      r0 = d0 * rc + d1 * rs;          // rotated[:,0]
      r1 = d1 * rc - d0 * rs;          // rotated[:,1]
    } else { r0 = xp.x; r1 = xp.y; }

    // ---- layer 1 (per-lane, packed f16): act feats 2j,2j+1 in p16[j]
    const v2h r0h = { (_Float16)r0, (_Float16)r0 };
    const v2h r1h = { (_Float16)r1, (_Float16)r1 };
    unsigned p16[16];
#pragma unroll
    for (int j = 0; j < 16; ++j) {
      v2h h = __builtin_bit_cast(v2h, w0p[j]) * r0h
            + __builtin_bit_cast(v2h, w1p[j]) * r1h
            + __builtin_bit_cast(v2h, bp[j]);
      p16[j] = lrelu_pk(__builtin_bit_cast(unsigned, h));
    }
    // ---- relayout per-lane act -> WMMA B tiles: pre-select payload,
    //      one full-EXEC v_permlanex16, post-select (no exec branches)
    V16U b0t, b1t;   // points 0-15 / 16-31
#pragma unroll
    for (int j = 0; j < 8; ++j) {
      unsigned send = lo ? p16[j + 8] : p16[j];
      unsigned recv = xor16u(send);
      b0t.u[j] = lo ? p16[j] : recv;
      b1t.u[j] = lo ? recv   : p16[j + 8];
    }

    // ---- two point-halves sequentially: halves peak D liveness, no spills
    process_half(b0t.v);
    process_half(b1t.v);
  }

  // ---- reduce over point columns (lanes) within each 16-lane half
  V8FU alo, ahi; alo.v = acc_lo; ahi.v = acc_hi;
#pragma unroll
  for (int off = 1; off < 16; off <<= 1) {
#pragma unroll
    for (int r = 0; r < 8; ++r) {
      alo.f[r] += __shfl_xor(alo.f[r], off, 32);
      ahi.f[r] += __shfl_xor(ahi.f[r], off, 32);
    }
  }
  __shared__ float red[4][32];
  if (lane == 0) {
#pragma unroll
    for (int r = 0; r < 8; ++r) { red[wave][r]      = alo.f[r]; red[wave][16 + r] = ahi.f[r]; }
  } else if (lane == 16) {
#pragma unroll
    for (int r = 0; r < 8; ++r) { red[wave][8 + r]  = alo.f[r]; red[wave][24 + r] = ahi.f[r]; }
  }
  __syncthreads();
  if (threadIdx.x < 32) {
    float s = red[0][threadIdx.x] + red[1][threadIdx.x] +
              red[2][threadIdx.x] + red[3][threadIdx.x];
    int slot = ROT ? ((b * NANCH + a) * NCHUNK + chunk) : (b * NCHUNK + chunk);
    partial[slot * 32 + threadIdx.x] = s;
  }
}

// tiny anchor-selector MLP [32->32->32->48] on 8 rows, plus cos/sin
__global__ void anchors_kernel(const float* __restrict__ p1, int nc1,
                               const float* __restrict__ aeB3,
                               const float* __restrict__ W1, const float* __restrict__ B1,
                               const float* __restrict__ W2, const float* __restrict__ B2,
                               const float* __restrict__ W3, const float* __restrict__ B3,
                               float4* __restrict__ anc)
{
  __shared__ float g[32], h1[32], h2[32], o[48];
  const int b = blockIdx.x, t = threadIdx.x;
  if (t < 32) {
    float s = 0.f;
    for (int c = 0; c < nc1; ++c) s += p1[(b * nc1 + c) * 32 + t];
    g[t] = s * (1.0f / POINTS) + aeB3[t];
  }
  __syncthreads();
  if (t < 32) {
    float s = B1[t];
    for (int k = 0; k < 32; ++k) s += g[k] * W1[t * 32 + k];
    h1[t] = fmaxf(s, 0.01f * s);
  }
  __syncthreads();
  if (t < 32) {
    float s = B2[t];
    for (int k = 0; k < 32; ++k) s += h1[k] * W2[t * 32 + k];
    h2[t] = fmaxf(s, 0.01f * s);
  }
  __syncthreads();
  if (t < 48) {
    float s = B3[t];
    for (int k = 0; k < 32; ++k) s += h2[k] * W3[t * 32 + k];
    o[t] = s;
  }
  __syncthreads();
  if (t < 16) {
    float ang = o[3 * t + 2];
    anc[b * NANCH + t] = make_float4(o[3 * t], o[3 * t + 1], cosf(ang), sinf(ang));
  }
}

// deterministic final mean; layer-3 bias folded in (mean of constant == constant)
__global__ void finalize_kernel(const float* __restrict__ p3, int nc3,
                                const float* __restrict__ peB3,
                                float* __restrict__ out)
{
  const int t = threadIdx.x;        // 256 = 8 batches * 32 feats
  const int b = t >> 5, f = t & 31;
  const int n = NANCH * nc3;
  float s = 0.f;
  for (int i = 0; i < n; ++i) s += p3[(b * n + i) * 32 + f];
  out[t] = peB3[f] + s * (1.0f / ((float)POINTS * (float)NANCH));
}

extern "C" void kernel_launch(void* const* d_in, const int* in_sizes, int n_in,
                              void* d_out, int out_size, void* d_ws, size_t ws_size,
                              hipStream_t stream) {
  (void)in_sizes; (void)n_in; (void)out_size; (void)ws_size;
  const float* x    = (const float*)d_in[0];
  const float* aeW1 = (const float*)d_in[1];
  const float* aeB1 = (const float*)d_in[2];
  const float* aeW2 = (const float*)d_in[3];
  const float* aeB2 = (const float*)d_in[4];
  const float* aeW3 = (const float*)d_in[5];
  const float* aeB3 = (const float*)d_in[6];
  const float* asW1 = (const float*)d_in[7];
  const float* asB1 = (const float*)d_in[8];
  const float* asW2 = (const float*)d_in[9];
  const float* asB2 = (const float*)d_in[10];
  const float* asW3 = (const float*)d_in[11];
  const float* asB3 = (const float*)d_in[12];
  const float* peW1 = (const float*)d_in[13];
  const float* peB1 = (const float*)d_in[14];
  const float* peW2 = (const float*)d_in[15];
  const float* peB2 = (const float*)d_in[16];
  const float* peW3 = (const float*)d_in[17];
  const float* peB3 = (const float*)d_in[18];
  float* out = (float*)d_out;

  char* ws = (char*)d_ws;
  float4* anc = (float4*)ws;                                     // 8*16*16B = 2 KB
  float*  p1  = (float*)(ws + 2048);                             // 8*NC1*32*4 = 16 KB
  float*  p3  = (float*)(ws + 2048 + NBATCH * NC1 * 32 * 4);     // 8*16*NC3*32*4 = 64 KB

  mlp3_kernel<false, NC1><<<NBATCH * NC1, 128, 0, stream>>>(
      (const float2*)x, aeW1, aeB1, aeW2, aeB2, aeW3, nullptr, p1);
  anchors_kernel<<<NBATCH, 64, 0, stream>>>(
      p1, NC1, aeB3, asW1, asB1, asW2, asB2, asW3, asB3, anc);
  mlp3_kernel<true, NC3><<<NBATCH * NANCH * NC3, 128, 0, stream>>>(
      (const float2*)x, peW1, peB1, peW2, peB2, peW3, anc, p3);
  finalize_kernel<<<1, 256, 0, stream>>>(p3, NC3, peB3, out);
}
